// NeuralRenderer_35287451304788
// MI455X (gfx1250) — matrix operand order repaired and checked by reading, not compiled, not run on hardware
//
#include <hip/hip_runtime.h>
#include <hip/hip_fp16.h>

#define B_RAYS 32768
#define NC 64
#define NF 32
#define NALL 96
#define HID 64

typedef __attribute__((ext_vector_type(16))) _Float16 v16h;
typedef __attribute__((ext_vector_type(8)))  float    v8f;

// ---------------------------------------------------------------------------
// Kernel 1: coarse pass, sigma only (matrix-vector: h . W2[:,3]) — VALU.
// thread = one coarse sample (B*64 threads)
// ---------------------------------------------------------------------------
__global__ __launch_bounds__(256) void nerf_coarse_sigma(
    const float* __restrict__ rays, const float* __restrict__ zrand,
    const float* __restrict__ W1,   const float* __restrict__ b1,
    const float* __restrict__ W2,   const float* __restrict__ b2,
    float* __restrict__ sigma_out)
{
    __shared__ float sW1[6 * HID];
    __shared__ float sb1[HID];
    __shared__ float sW2c[HID];   // W2[:,3]

    int t = threadIdx.x;
    for (int i = t; i < 6 * HID; i += 256) sW1[i] = W1[i];
    if (t < HID) { sb1[t] = b1[t]; sW2c[t] = W2[t * HID + 3]; }
    __syncthreads();

    int gid = blockIdx.x * 256 + t;      // = ray*64 + i
    int r = gid >> 6, i = gid & 63;
    const float* ray = rays + r * 8;
    float ox = ray[0], oy = ray[1], oz = ray[2];
    float dx = ray[3], dy = ray[4], dz = ray[5];
    float nv = ray[6], fv = ray[7];

    float s = ((float)i + zrand[gid]) * (1.0f / NC);   // linspace step == 1/NC
    float z = nv * (1.0f - s) + fv * s;
    float x0 = fmaf(z, dx, ox), x1 = fmaf(z, dy, oy), x2 = fmaf(z, dz, oz);

    float sg = b2[3];
    #pragma unroll 4
    for (int j = 0; j < HID; j++) {
        float h = sb1[j];
        h = fmaf(x0, sW1[0 * HID + j], h);
        h = fmaf(x1, sW1[1 * HID + j], h);
        h = fmaf(x2, sW1[2 * HID + j], h);
        h = fmaf(dx, sW1[3 * HID + j], h);
        h = fmaf(dy, sW1[4 * HID + j], h);
        h = fmaf(dz, sW1[5 * HID + j], h);
        sg = fmaf(fmaxf(h, 0.0f), sW2c[j], sg);
    }
    sigma_out[gid] = sg;
}

// ---------------------------------------------------------------------------
// Kernel 2: per-ray weights -> CDF -> inverse-CDF fine samples -> sorted merge
// thread = one ray, per-thread arrays in LDS (block 64 => 40KB LDS)
// ---------------------------------------------------------------------------
__global__ __launch_bounds__(64) void nerf_fine_sample(
    const float* __restrict__ rays,  const float* __restrict__ zrand,
    const float* __restrict__ ufine, const float* __restrict__ jit,
    const float* __restrict__ sigma_ws, float* __restrict__ zall_ws)
{
    __shared__ float scdf[64][NC];
    __shared__ float szl[64][NALL];

    int t = threadIdx.x;
    int r = blockIdx.x * 64 + t;
    const float* ray = rays + r * 8;
    float nv = ray[6], fv = ray[7];
    float* cdf = scdf[t];
    float* zl  = szl[t];

    for (int i = 0; i < NC; i++) {
        float s = ((float)i + zrand[r * NC + i]) * (1.0f / NC);
        zl[i] = nv * (1.0f - s) + fv * s;
    }
    // weights (exclusive transmittance) + unnormalized cdf
    float T = 1.0f, c = 0.0f;
    for (int i = 0; i < NC; i++) {
        float zi = zl[i];
        float zn = (i < NC - 1) ? zl[i + 1] : fv;
        float sg = fmaxf(sigma_ws[r * NC + i], 0.0f);
        float alpha = 1.0f - __expf(-(zn - zi) * sg);
        float w = alpha * T + 1e-5f;
        T *= (1.0f - alpha + 1e-10f);
        c += w;
        cdf[i] = c;
    }
    float S = c;
    // inverse CDF: ind = #{j : cdf[j] <= u*S}  ( == searchsorted(right)-1 )
    for (int k = 0; k < NF; k++) {
        float target = ufine[r * NF + k] * S;
        int ind = 0;
        for (int j = 0; j < NC; j++) ind += (cdf[j] <= target) ? 1 : 0;
        float sf = ((float)ind + jit[r * NF + k]) * (1.0f / NC);
        float zf = nv * (1.0f - sf) + fv * sf;
        // insertion-merge into sorted zl (coarse z already sorted)
        int len = NC + k;
        int pos = 0;
        while (pos < len && zl[pos] <= zf) pos++;
        for (int q = len; q > pos; q--) zl[q] = zl[q - 1];
        zl[pos] = zf;
    }
    for (int i = 0; i < NALL; i++) zall_ws[r * NALL + i] = zl[i];
}

// ---------------------------------------------------------------------------
// Kernel 3: final pass. 4 waves/block, one wave = one ray.
// 96 samples = 6 M-tiles of 16; layer-2 via v_wmma_f32_16x16x32_f16.
// A frag (16x32 f16): lane L holds row M=L%16, k = (i<8?i:i+8)+8*(L/16)
// B frag (32x16 f16): lane L holds col N=L%16, k = (i<8?i:i+8)+8*(L/16)
// D (16x16 f32): VGPR rr -> M = rr+8*(L/16), N = L%16
// ---------------------------------------------------------------------------
__global__ __launch_bounds__(128) void nerf_final(
    const float* __restrict__ rays, const float* __restrict__ zall_ws,
    const float* __restrict__ W1,   const float* __restrict__ b1,
    const float* __restrict__ W2,   const float* __restrict__ b2,
    float* __restrict__ out)
{
    __shared__ float     sW1[6 * HID];
    __shared__ float     sb1[HID];
    __shared__ float     sb2[HID];
    __shared__ _Float16  sW2h[HID * HID];
    __shared__ float     szall[4][NALL];
    __shared__ float     soutb[4][16][HID];
    __shared__ float     sg[4][16];
    __shared__ float     swt[4][16];

    int tid  = threadIdx.x;
    int wid  = tid >> 5, lane = tid & 31;
    int lmod = lane & 15, lhi = lane >> 4;
    int r = blockIdx.x * 4 + wid;

    for (int i = tid; i < 6 * HID; i += 128) sW1[i] = W1[i];
    for (int i = tid; i < HID; i += 128) { sb1[i] = b1[i]; sb2[i] = b2[i]; }
    for (int i = tid; i < HID * HID; i += 128) sW2h[i] = (_Float16)W2[i];
    for (int i = lane; i < NALL; i += 32) szall[wid][i] = zall_ws[r * NALL + i];
    __syncthreads();

    const float* ray = rays + r * 8;
    float ox = ray[0], oy = ray[1], oz = ray[2];
    float dx = ray[3], dy = ray[4], dz = ray[5];
    float fv = ray[7];

    // B fragments of W2 (kept in registers for all 6 tiles)
    v16h bfrag[2][4];
    #pragma unroll
    for (int kh = 0; kh < 2; kh++)
        #pragma unroll
        for (int n = 0; n < 4; n++)
            #pragma unroll
            for (int i = 0; i < 16; i++) {
                int k = ((i < 8) ? i : (i + 8)) + 8 * lhi + 32 * kh;          // hidden index
                bfrag[kh][n][i] = sW2h[k * HID + n * 16 + lmod];
            }

    float T = 1.0f, acc1 = 0.0f, acc2 = 0.0f, accd = 0.0f;

    for (int t = 0; t < 6; t++) {
        // ---- layer 1: each lane computes exactly its A-fragment elements ----
        float z = szall[wid][t * 16 + lmod];
        float xin[6];
        xin[0] = fmaf(z, dx, ox); xin[1] = fmaf(z, dy, oy); xin[2] = fmaf(z, dz, oz);
        xin[3] = dx; xin[4] = dy; xin[5] = dz;

        v16h afrag[2];
        #pragma unroll
        for (int kh = 0; kh < 2; kh++)
            #pragma unroll
            for (int i = 0; i < 16; i++) {
                int j = ((i < 8) ? i : (i + 8)) + 8 * lhi + 32 * kh;
                float h = sb1[j];
                #pragma unroll
                for (int c6 = 0; c6 < 6; c6++) h = fmaf(xin[c6], sW1[c6 * HID + j], h);
                afrag[kh][i] = (_Float16)fmaxf(h, 0.0f);
            }

        // ---- layer 2: 4 N-tiles x 2 K-chained WMMAs ----
        #pragma unroll
        for (int n = 0; n < 4; n++) {
            v8f d = {0.f, 0.f, 0.f, 0.f, 0.f, 0.f, 0.f, 0.f};
            d = __builtin_amdgcn_wmma_f32_16x16x32_f16(
                    false, afrag[0], false, bfrag[0][n], (short)0, d, false, false);
            d = __builtin_amdgcn_wmma_f32_16x16x32_f16(
                    false, afrag[1], false, bfrag[1][n], (short)0, d, false, false);
            int col = n * 16 + lmod;
            #pragma unroll
            for (int rr = 0; rr < 8; rr++)
                soutb[wid][rr + 8 * lhi][col] = d[rr] + sb2[col];
        }
        __syncthreads();

        // ---- alphas for this tile (lanes 0-15 own samples) ----
        {
            int sidx = t * 16 + lmod;
            float sig = soutb[wid][lmod][3];
            float zs = szall[wid][sidx];
            float zn = (sidx == NALL - 1) ? fv : szall[wid][sidx + 1];
            float alpha = 1.0f - __expf(-(zn - zs) * fmaxf(sig, 0.0f));
            if (lane < 16) sg[wid][lane] = alpha;
        }
        __syncthreads();

        // ---- wave-uniform exclusive transmittance scan over 16 samples ----
        float prod = 1.0f, myw = 0.0f;
        #pragma unroll
        for (int k = 0; k < 16; k++) {
            float a = sg[wid][k];
            if (k == lmod) myw = a * T * prod;
            prod *= (1.0f - a + 1e-10f);
        }
        if (lane < 16) swt[wid][lane] = myw;
        T *= prod;                       // uniform update, carries across tiles
        __syncthreads();

        // ---- weighted accumulation: lane handles cols (lane, lane+32) ----
        for (int mm = 0; mm < 16; mm++) {
            float w  = swt[wid][mm];
            float v1 = soutb[wid][mm][lane];
            float v2 = soutb[wid][mm][lane + 32];
            if (lane < 3) v1 = 1.0f / (1.0f + __expf(-v1));   // sigmoid rgb
            acc1 += w * v1;
            acc2 += w * v2;
            accd += w * szall[wid][t * 16 + mm];              // depth (uniform)
        }
        __syncthreads();
    }

    // output layout: [rgb(0..2), embed = cols 4..63 -> 3..62, depth -> 63]
    float* orow = out + r * 64;
    if (lane < 3)       orow[lane]     = acc1;
    else if (lane == 3) orow[63]       = accd;   // col 3 (sigma) excluded
    else                orow[lane - 1] = acc1;
    orow[lane + 31] = acc2;                      // cols 32..63 -> 31..62
}

// ---------------------------------------------------------------------------
extern "C" void kernel_launch(void* const* d_in, const int* in_sizes, int n_in,
                              void* d_out, int out_size, void* d_ws, size_t ws_size,
                              hipStream_t stream) {
    (void)in_sizes; (void)n_in; (void)out_size; (void)ws_size;
    const float* rays  = (const float*)d_in[0];
    const float* zrand = (const float*)d_in[1];
    const float* ufine = (const float*)d_in[2];
    const float* jit   = (const float*)d_in[3];
    const float* W1    = (const float*)d_in[4];
    const float* b1    = (const float*)d_in[5];
    const float* W2    = (const float*)d_in[6];
    const float* b2    = (const float*)d_in[7];
    float* out = (float*)d_out;

    float* sigma_ws = (float*)d_ws;                 // B*64 f32  (8 MB)
    float* zall_ws  = sigma_ws + B_RAYS * NC;       // B*96 f32  (12 MB)

    nerf_coarse_sigma<<<(B_RAYS * NC) / 256, 256, 0, stream>>>(
        rays, zrand, W1, b1, W2, b2, sigma_ws);
    nerf_fine_sample<<<B_RAYS / 64, 64, 0, stream>>>(
        rays, zrand, ufine, jit, sigma_ws, zall_ws);
    nerf_final<<<B_RAYS / 4, 128, 0, stream>>>(
        rays, zall_ws, W1, b1, W2, b2, out);
}
